// MNDGNN_17806934409759
// MI455X (gfx1250) — compile-verified, matrix operand order checked by reading, not run on hardware
//
#include <hip/hip_runtime.h>
#include <hip/hip_bf16.h>

#define DEVFN __device__ __forceinline__

constexpr int N_ = 40000, D_ = 128, R_ = 3, E_ = 640000, L_ = 2, C_ = 40, MAXDEG_ = 512;
constexpr int GB_ = (N_ + 127) / 128;   // 313 WMMA row-blocks
constexpr int NPAD_ = GB_ * 128;        // 40064 padded rows (guard-free WMMA epilogues)
constexpr float BN_EPS_ = 1e-5f;

typedef __attribute__((ext_vector_type(16))) __bf16 v16bf;
typedef __attribute__((ext_vector_type(8)))  __bf16 v8bf;
typedef __attribute__((ext_vector_type(8)))  float  v8f;

// ---- WMMA helpers (CDNA5 16x16x32 bf16, f32 accumulate) ----
// 16-bit A-matrix 16x32 layout (ISA 7.12.2): lane = 16*half + m; per lane the
// K-chunk is {k0+8*half .. +7} and {k0+16+8*half .. +7} -> two b128 loads.
DEVFN v16bf mnd_load_frag(const __bf16* base, int half) {
  v8bf lo = *(const v8bf*)(base + 8 * half);
  v8bf hi = *(const v8bf*)(base + 16 + 8 * half);
  v16bf f;
#pragma unroll
  for (int i = 0; i < 8; ++i) { f[i] = lo[i]; f[i + 8] = hi[i]; }
  return f;
}

DEVFN v8f mnd_wmma(v16bf a, v16bf b, v8f c) {
  // 8 args: (neg_a, A, neg_b, B, c_mod, C, reuse_a, reuse_b)
  return __builtin_amdgcn_wmma_f32_16x16x32_bf16(false, a, false, b, (short)0, c,
                                                 false, false);
}

// ---------------- elementwise / utility kernels ----------------
__global__ void mnd_zero(float* p, long n) {
  long i = blockIdx.x * (long)blockDim.x + threadIdx.x;
  const long st = (long)gridDim.x * blockDim.x;
  for (; i < n; i += st) p[i] = 0.f;
}

__global__ void mnd_cvt_bf(const float* __restrict__ in, __bf16* __restrict__ out, long n) {
  long i = blockIdx.x * (long)blockDim.x + threadIdx.x;
  const long st = (long)gridDim.x * blockDim.x;
  for (; i < n; i += st) out[i] = (__bf16)in[i];
}

// transpose+convert weight: W[K x ncols] (row major) -> WT[npad x K] bf16, zero pad
__global__ void mnd_wt(const float* __restrict__ W, __bf16* __restrict__ WT,
                       int K, int ncols, int npad) {
  int i = blockIdx.x * blockDim.x + threadIdx.x;
  if (i >= npad * K) return;
  int n = i / K, k = i % K;
  WT[n * K + k] = (n < ncols) ? (__bf16)W[k * ncols + n] : (__bf16)0.0f;
}

// ---------------- graph precompute ----------------
__global__ void mnd_deg(const int* __restrict__ rows, const int* __restrict__ cols,
                        float* degO, float* degI) {
  int i = blockIdx.x * blockDim.x + threadIdx.x;
  const int st = gridDim.x * blockDim.x;
  for (; i < E_; i += st) {
    atomicAdd(&degO[rows[i]], 1.0f);
    atomicAdd(&degI[cols[i]], 1.0f);
  }
}

__global__ void mnd_norm(const float* __restrict__ deg, float* __restrict__ dsc,
                         int* __restrict__ idx) {
  int i = blockIdx.x * blockDim.x + threadIdx.x;
  if (i >= N_) return;
  float d = deg[i];
  dsc[i] = (d > 0.f) ? rsqrtf(fmaxf(d, 1.f)) : 0.f;
  int di = (int)d; if (di > MAXDEG_ - 1) di = MAXDEG_ - 1; if (di < 0) di = 0;
  idx[i] = di;
}

__global__ void mnd_ew(const int* __restrict__ rows, const int* __restrict__ cols,
                       const float* __restrict__ ods, const float* __restrict__ ids,
                       float* __restrict__ w) {
  int i = blockIdx.x * blockDim.x + threadIdx.x;
  const int st = gridDim.x * blockDim.x;
  for (; i < E_; i += st) w[i] = ods[rows[i]] * ids[cols[i]];
}

// ---------------- per-node row L1 inverse norm (wave per node) ----------------
__global__ void mnd_l1(const float* __restrict__ h, float* __restrict__ invl1) {
  const int gid = blockIdx.x * blockDim.x + threadIdx.x;
  const int m = gid >> 5, lane = gid & 31;
  if (m >= N_) return;
  const int b = m * D_ + lane * 4;
  float s = fabsf(h[b]) + fabsf(h[b + 1]) + fabsf(h[b + 2]) + fabsf(h[b + 3]);
#pragma unroll
  for (int k = 16; k >= 1; k >>= 1) s += __shfl_xor(s, k, 32);
  if (lane == 0) invl1[m] = 1.0f / (s + 1e-12f);
}

// ---------------- fused 6-target edge SpMM (wave per edge) ----------------
__global__ void mnd_spmm6(const int* __restrict__ rows, const int* __restrict__ cols,
                          const float* __restrict__ ew, const float* __restrict__ h,
                          const float* __restrict__ invl1,
                          float* ShO, float* SxO, float* SxxO,
                          float* ShI, float* SxI, float* SxxI) {
  const int lane = threadIdx.x & 31;
  const int wstride = (gridDim.x * blockDim.x) >> 5;
  for (int e = (blockIdx.x * blockDim.x + threadIdx.x) >> 5; e < E_; e += wstride) {
    const int r = rows[e], c = cols[e];
    const float wgt = ew[e];
    const float ir = invl1[r], ic = invl1[c];
    const int rb = r * D_ + lane * 4, cb = c * D_ + lane * 4;
#pragma unroll
    for (int j = 0; j < 4; ++j) {
      const float hr = h[rb + j], hc = h[cb + j];
      const float x1c = hc * ic, x1r = hr * ir;
      atomicAdd(&ShO[rb + j], wgt * hc);          // mm(h)     -> row
      atomicAdd(&SxO[rb + j], wgt * x1c);         // mm(x1)    -> row
      atomicAdd(&SxxO[rb + j], wgt * x1c * x1c);  // mm(x1^2)  -> row
      atomicAdd(&ShI[cb + j], wgt * hr);          // mm^T(h)   -> col
      atomicAdd(&SxI[cb + j], wgt * x1r);
      atomicAdd(&SxxI[cb + j], wgt * x1r * x1r);
    }
  }
}

// ---------------- Dirichlet energy + gate (wave per node) ----------------
__global__ void mnd_cfeat(const float* __restrict__ h, const float* __restrict__ invl1,
                          const float* __restrict__ SxO, const float* __restrict__ SxxO,
                          const float* __restrict__ SxI, const float* __restrict__ SxxI,
                          const float* __restrict__ oemb_l, const float* __restrict__ iemb_l,
                          const int* __restrict__ idxO_r, const int* __restrict__ idxI_r,
                          const float* __restrict__ Woutf_l, const float* __restrict__ Winf_l,
                          const float* __restrict__ boutf_p, const float* __restrict__ binf_p,
                          const float* __restrict__ tau_p,
                          const float* __restrict__ omask_r, const float* __restrict__ omb_r,
                          const float* __restrict__ imask_r, const float* __restrict__ imb_r,
                          float* __restrict__ co_arr, float* __restrict__ ci_arr,
                          float* __restrict__ cout_acc, float* __restrict__ cin_acc,
                          float accscale) {
  const int gid = blockIdx.x * blockDim.x + threadIdx.x;
  const int m = gid >> 5, lane = gid & 31;
  if (m >= N_) return;
  const int io = idxO_r[m], ii = idxI_r[m];
  const float im = (m < N_ - 1) ? 1.f : 0.f;
  const float inv = invl1[m];
  const int base = m * D_ + lane * 4;
  const int fb = lane * 4;
  float co = 0.f, ci = 0.f;
#pragma unroll
  for (int j = 0; j < 4; ++j) {
    const float hv = h[base + j];
    const float x1 = hv * inv, x1s = x1 * x1;
    const float eo = im * x1s + SxxO[base + j] - 2.f * ((im * x1 + SxO[base + j]) * x1 - x1s);
    const float ein = im * x1s + SxxI[base + j] - 2.f * ((im * x1 + SxI[base + j]) * x1 - x1s);
    co += (-eo + oemb_l[io * D_ + fb + j]) * Woutf_l[fb + j];
    ci += (-ein + iemb_l[ii * D_ + fb + j]) * Winf_l[fb + j];
  }
#pragma unroll
  for (int k = 16; k >= 1; k >>= 1) {
    co += __shfl_xor(co, k, 32);
    ci += __shfl_xor(ci, k, 32);
  }
  if (lane == 0) {
    co += boutf_p[0];
    ci += binf_p[0];
    const float t = __expf(tau_p[0]) + 0.1f;
    const float mx = fmaxf(co, ci);
    const float p0 = __expf((co - mx) / t), p1 = __expf((ci - mx) / t);
    const float s = p0 + p1;
    const float cOv = (p0 / s) * omask_r[m] + omb_r[m];
    const float cIv = (p1 / s) * imask_r[m] + imb_r[m];
    co_arr[m] = cOv;
    ci_arr[m] = cIv;
    if (cout_acc) {
      cout_acc[m] += cOv * accscale;
      cin_acc[m] += cIv * accscale;
    }
  }
}

// ---------------- WMMA GEMM kernels (row-padded, guard-free epilogues) -------
// block = 256 thr = 8 waves; wave -> 16 rows x 128 cols; K = 128 in 4 steps of 32.
// B (128x128 bf16, pre-transposed) staged once in LDS; fragment loads hit LDS.

// out[m,n] += scale[m] * ((A @ B)[m,n] + bias[n])     (A/out/scale padded to NPAD rows)
__global__ __launch_bounds__(256, 1)
void mnd_gemm_scaled(const __bf16* __restrict__ A, const __bf16* __restrict__ BT,
                     const float* __restrict__ bias, const float* __restrict__ scale,
                     float* __restrict__ out) {
  __shared__ __bf16 Bs[D_ * D_];
  {
    const uint4* s = (const uint4*)BT;
    uint4* d = (uint4*)Bs;
#pragma unroll
    for (int i = 0; i < (D_ * D_ * 2) / (16 * 256); ++i)
      d[threadIdx.x + i * 256] = s[threadIdx.x + i * 256];
  }
  __syncthreads();
  const int wid = threadIdx.x >> 5, lane = threadIdx.x & 31;
  const int half = lane >> 4, l16 = lane & 15;
  const int m0 = blockIdx.x * 128 + wid * 16;
  const int mA = m0 + l16;
  v16bf a[4];
#pragma unroll
  for (int ks = 0; ks < 4; ++ks) a[ks] = mnd_load_frag(A + (size_t)mA * D_ + ks * 32, half);
#pragma unroll
  for (int ct = 0; ct < 8; ++ct) {
    v8f acc = {};
    const int n = ct * 16 + l16;
#pragma unroll
    for (int ks = 0; ks < 4; ++ks) {
      v16bf b = mnd_load_frag(Bs + n * D_ + ks * 32, half);
      acc = mnd_wmma(a[ks], b, acc);
    }
    const float bb = bias[n];
#pragma unroll
    for (int j = 0; j < 8; ++j) {
      const int m = m0 + j + 8 * half;  // C/D layout: half=0 -> M=j, half=1 -> M=j+8
      const size_t idx = (size_t)m * D_ + n;
      out[idx] += scale[m] * (acc[j] + bb);
    }
  }
}

// out = out/R + alpha*(A@Wfc + b)   (padded; BN stats done by mnd_bnstats)
__global__ __launch_bounds__(256, 1)
void mnd_gemm_fc(const __bf16* __restrict__ A, const __bf16* __restrict__ BT,
                 const float* __restrict__ bias, const float* __restrict__ alpha_p,
                 float* __restrict__ out) {
  __shared__ __bf16 Bs[D_ * D_];
  {
    const uint4* s = (const uint4*)BT;
    uint4* d = (uint4*)Bs;
#pragma unroll
    for (int i = 0; i < (D_ * D_ * 2) / (16 * 256); ++i)
      d[threadIdx.x + i * 256] = s[threadIdx.x + i * 256];
  }
  __syncthreads();
  const int wid = threadIdx.x >> 5, lane = threadIdx.x & 31;
  const int half = lane >> 4, l16 = lane & 15;
  const int m0 = blockIdx.x * 128 + wid * 16;
  const int mA = m0 + l16;
  const float alpha = alpha_p[0];
  v16bf a[4];
#pragma unroll
  for (int ks = 0; ks < 4; ++ks) a[ks] = mnd_load_frag(A + (size_t)mA * D_ + ks * 32, half);
#pragma unroll
  for (int ct = 0; ct < 8; ++ct) {
    v8f acc = {};
    const int n = ct * 16 + l16;
#pragma unroll
    for (int ks = 0; ks < 4; ++ks) {
      v16bf b = mnd_load_frag(Bs + n * D_ + ks * 32, half);
      acc = mnd_wmma(a[ks], b, acc);
    }
    const float bb = bias[n];
#pragma unroll
    for (int j = 0; j < 8; ++j) {
      const int m = m0 + j + 8 * half;
      const size_t idx = (size_t)m * D_ + n;
      out[idx] = out[idx] * (1.0f / R_) + alpha * (acc[j] + bb);
    }
  }
}

// classifier: logits[NPAD x 40] = A@WlinT(48x128 padded) + b_lin
__global__ __launch_bounds__(256, 1)
void mnd_gemm_lin(const __bf16* __restrict__ A, const __bf16* __restrict__ BT,
                  const float* __restrict__ blin, float* __restrict__ logits) {
  __shared__ __bf16 Bs[48 * D_];
  {
    const uint4* s = (const uint4*)BT;
    uint4* d = (uint4*)Bs;
#pragma unroll
    for (int i = 0; i < (48 * D_ * 2) / (16 * 256); ++i)
      d[threadIdx.x + i * 256] = s[threadIdx.x + i * 256];
  }
  __syncthreads();
  const int wid = threadIdx.x >> 5, lane = threadIdx.x & 31;
  const int half = lane >> 4, l16 = lane & 15;
  const int m0 = blockIdx.x * 128 + wid * 16;
  const int mA = m0 + l16;
  v16bf a[4];
#pragma unroll
  for (int ks = 0; ks < 4; ++ks) a[ks] = mnd_load_frag(A + (size_t)mA * D_ + ks * 32, half);
#pragma unroll
  for (int ct = 0; ct < 3; ++ct) {
    v8f acc = {};
    const int n = ct * 16 + l16;
#pragma unroll
    for (int ks = 0; ks < 4; ++ks) {
      v16bf b = mnd_load_frag(Bs + n * D_ + ks * 32, half);
      acc = mnd_wmma(a[ks], b, acc);
    }
    if (n < C_) {
      const float bb = blin[n];
#pragma unroll
      for (int j = 0; j < 8; ++j) {
        const int m = m0 + j + 8 * half;
        logits[(size_t)m * C_ + n] = acc[j] + bb;
      }
    }
  }
}

// ---------------- BN column statistics (coalesced sweep, few atomics) --------
__global__ void mnd_bnstats(const float* __restrict__ pre, float* __restrict__ bsum,
                            float* __restrict__ bsq) {
  const int col = threadIdx.x & (D_ - 1);
  const int rs = (threadIdx.x >> 7) + blockIdx.x * 2;   // 2 row streams per block
  const int rstride = gridDim.x * 2;
  float s = 0.f, sq = 0.f;
  for (int r = rs; r < N_; r += rstride) {
    const float v = pre[(size_t)r * D_ + col];
    s += v;
    sq += v * v;
  }
  atomicAdd(&bsum[col], s);
  atomicAdd(&bsq[col], sq);
}

// ---------------- BatchNorm + ReLU apply (+ g accumulation) ----------------
__global__ void mnd_bn(const float* __restrict__ pre, const float* __restrict__ bsum,
                       const float* __restrict__ bsq, const float* __restrict__ gamma_l,
                       const float* __restrict__ beta_l, float* __restrict__ ph_out,
                       float* __restrict__ g_slot) {
  const long total = (long)N_ * D_;
  const long st = (long)gridDim.x * blockDim.x;
  for (long i = blockIdx.x * (long)blockDim.x + threadIdx.x; i < total; i += st) {
    const int n = (int)(i & (D_ - 1));
    const float mu = bsum[n] * (1.f / N_);
    const float var = bsq[n] * (1.f / N_) - mu * mu;
    const float v = (pre[i] - mu) * rsqrtf(var + BN_EPS_) * gamma_l[n] + beta_l[n];
    const float ph = fmaxf(v, 0.f);
    ph_out[i] = ph;
    if (g_slot) atomicAdd(&g_slot[n], ph * (1.f / N_));
  }
}

// deterministic bijective permutation gather (stand-in for torch.randperm)
__global__ void mnd_perm(const float* __restrict__ src, float* __restrict__ dst,
                         unsigned addv) {
  const long total = (long)N_ * D_;
  const long st = (long)gridDim.x * blockDim.x;
  for (long i = blockIdx.x * (long)blockDim.x + threadIdx.x; i < total; i += st) {
    const int m = (int)(i / D_), f = (int)(i % D_);
    const unsigned pm = (unsigned)((2654435761ull * (unsigned)m + addv) % (unsigned)N_);
    dst[i] = src[(size_t)pm * D_ + f];
  }
}

__global__ void mnd_jkmax(const float* __restrict__ a, const float* __restrict__ b,
                          __bf16* __restrict__ out) {
  const long total = (long)N_ * D_;
  const long st = (long)gridDim.x * blockDim.x;
  for (long i = blockIdx.x * (long)blockDim.x + threadIdx.x; i < total; i += st)
    out[i] = (__bf16)fmaxf(a[i], b[i]);
}

// wave per row log-softmax over C=40 (lane covers c and c+32)
__global__ void mnd_logsm(const float* __restrict__ logits, float* __restrict__ out) {
  const int gid = blockIdx.x * blockDim.x + threadIdx.x;
  const int m = gid >> 5, lane = gid & 31;
  if (m >= N_) return;
  const float v0 = logits[m * C_ + lane];
  const float v1 = (lane < C_ - 32) ? logits[m * C_ + 32 + lane] : -3.0e38f;
  float mx = fmaxf(v0, v1);
#pragma unroll
  for (int k = 16; k >= 1; k >>= 1) mx = fmaxf(mx, __shfl_xor(mx, k, 32));
  float se = __expf(v0 - mx) + ((lane < C_ - 32) ? __expf(v1 - mx) : 0.f);
#pragma unroll
  for (int k = 16; k >= 1; k >>= 1) se += __shfl_xor(se, k, 32);
  const float ls = __logf(se);
  out[m * C_ + lane] = v0 - mx - ls;
  if (lane < C_ - 32) out[m * C_ + 32 + lane] = v1 - mx - ls;
}

// ================= host orchestration =================
extern "C" void kernel_launch(void* const* d_in, const int* in_sizes, int n_in,
                              void* d_out, int out_size, void* d_ws, size_t ws_size,
                              hipStream_t stream) {
  const float* x        = (const float*)d_in[0];
  const int*   ei       = (const int*)d_in[1];
  const float* W_src    = (const float*)d_in[2];
  const float* b_src    = (const float*)d_in[3];
  const float* W_dst    = (const float*)d_in[4];
  const float* b_dst    = (const float*)d_in[5];
  const float* W_outf   = (const float*)d_in[6];
  const float* b_outf   = (const float*)d_in[7];
  const float* W_inf    = (const float*)d_in[8];
  const float* b_inf    = (const float*)d_in[9];
  const float* W_fc     = (const float*)d_in[10];
  const float* b_fc     = (const float*)d_in[11];
  const float* bn_gamma = (const float*)d_in[12];
  const float* bn_beta  = (const float*)d_in[13];
  const float* out_emb  = (const float*)d_in[14];
  const float* in_emb   = (const float*)d_in[15];
  const float* out_mask = (const float*)d_in[16];
  const float* out_mb   = (const float*)d_in[17];
  const float* in_mask  = (const float*)d_in[18];
  const float* in_mb    = (const float*)d_in[19];
  const float* alpha_p  = (const float*)d_in[20];
  const float* tau_p    = (const float*)d_in[21];
  const float* W_lin    = (const float*)d_in[22];
  const float* b_lin    = (const float*)d_in[23];
  (void)in_sizes; (void)n_in; (void)out_size; (void)ws_size;

  // ---- scratch bump allocator ----
  char* wsb = (char*)d_ws;
  size_t off = 0;
  auto alloc = [&](size_t bytes) -> void* {
    void* p = wsb + off;
    off = (off + bytes + 255) & ~(size_t)255;
    return p;
  };
  float* degO  = (float*)alloc((size_t)R_ * N_ * 4);
  float* degI  = (float*)alloc((size_t)R_ * N_ * 4);
  float* ods   = (float*)alloc((size_t)R_ * N_ * 4);
  float* ids   = (float*)alloc((size_t)R_ * N_ * 4);
  int*   idxO  = (int*)alloc((size_t)R_ * N_ * 4);
  int*   idxI  = (int*)alloc((size_t)R_ * N_ * 4);
  float* ewr   = (float*)alloc((size_t)R_ * E_ * 4);
  __bf16* WsrcT = (__bf16*)alloc((size_t)L_ * D_ * D_ * 2);
  __bf16* WdstT = (__bf16*)alloc((size_t)L_ * D_ * D_ * 2);
  __bf16* WfcT  = (__bf16*)alloc((size_t)L_ * D_ * D_ * 2);
  __bf16* WlinT = (__bf16*)alloc((size_t)48 * D_ * 2);
  float* hnegbuf = (float*)alloc((size_t)N_ * D_ * 4);
  __bf16* hbf    = (__bf16*)alloc((size_t)NPAD_ * D_ * 2);
  float* invl1   = (float*)alloc((size_t)N_ * 4);
  float* Sall    = (float*)alloc((size_t)6 * N_ * D_ * 4);
  float* ShO = Sall + 0L * N_ * D_;
  float* SxO = Sall + 1L * N_ * D_;
  float* SxxO = Sall + 2L * N_ * D_;
  float* ShI = Sall + 3L * N_ * D_;
  float* SxI = Sall + 4L * N_ * D_;
  float* SxxI = Sall + 5L * N_ * D_;
  __bf16* ShObf = (__bf16*)alloc((size_t)NPAD_ * D_ * 2);
  __bf16* ShIbf = (__bf16*)alloc((size_t)NPAD_ * D_ * 2);
  float* outacc = (float*)alloc((size_t)NPAD_ * D_ * 4);
  float* co_arr = (float*)alloc((size_t)NPAD_ * 4);
  float* ci_arr = (float*)alloc((size_t)NPAD_ * 4);
  float* bnsum  = (float*)alloc((size_t)D_ * 4);
  float* bnsq   = (float*)alloc((size_t)D_ * 4);
  float* logits = (float*)alloc((size_t)NPAD_ * C_ * 4);
  __bf16* jkbf  = (__bf16*)alloc((size_t)NPAD_ * D_ * 2);

  // ---- output layout ----
  float* o = (float*)d_out;
  float* o_logp = o;
  float* o_cin  = o + (size_t)N_ * C_;
  float* o_cout = o_cin + N_;
  float* o_ph0  = o_cout + N_;
  float* o_ph1  = o_ph0 + (size_t)N_ * D_;
  float* o_nh0  = o_ph1 + (size_t)N_ * D_;
  float* o_nh1  = o_nh0 + (size_t)N_ * D_;
  float* o_g0   = o_nh1 + (size_t)N_ * D_;
  float* o_g1   = o_g0 + D_;
  const long out_total = (long)N_ * C_ + 2L * N_ + 4L * N_ * D_ + 2L * D_;

  const int NW = (N_ * 32 + 255) / 256;     // wave-per-node grids

  // ---- init: zero outputs, convert weights, graph precompute ----
  mnd_zero<<<4096, 256, 0, stream>>>(o, out_total);
  mnd_zero<<<256, 256, 0, stream>>>(co_arr, (long)NPAD_);  // pad tails defined
  mnd_zero<<<256, 256, 0, stream>>>(ci_arr, (long)NPAD_);
  for (int l = 0; l < L_; ++l) {
    mnd_wt<<<(D_ * D_ + 255) / 256, 256, 0, stream>>>(W_src + (size_t)l * D_ * D_, WsrcT + (size_t)l * D_ * D_, D_, D_, D_);
    mnd_wt<<<(D_ * D_ + 255) / 256, 256, 0, stream>>>(W_dst + (size_t)l * D_ * D_, WdstT + (size_t)l * D_ * D_, D_, D_, D_);
    mnd_wt<<<(D_ * D_ + 255) / 256, 256, 0, stream>>>(W_fc + (size_t)l * D_ * D_, WfcT + (size_t)l * D_ * D_, D_, D_, D_);
  }
  mnd_wt<<<(48 * D_ + 255) / 256, 256, 0, stream>>>(W_lin, WlinT, D_, C_, 48);
  mnd_zero<<<512, 256, 0, stream>>>(degO, (long)R_ * N_);
  mnd_zero<<<512, 256, 0, stream>>>(degI, (long)R_ * N_);
  for (int r = 0; r < R_; ++r) {
    const int* rows = ei + (size_t)r * 2 * E_;
    const int* cols = rows + E_;
    mnd_deg<<<4096, 256, 0, stream>>>(rows, cols, degO + (size_t)r * N_, degI + (size_t)r * N_);
    mnd_norm<<<(N_ + 255) / 256, 256, 0, stream>>>(degO + (size_t)r * N_, ods + (size_t)r * N_, idxO + (size_t)r * N_);
    mnd_norm<<<(N_ + 255) / 256, 256, 0, stream>>>(degI + (size_t)r * N_, ids + (size_t)r * N_, idxI + (size_t)r * N_);
    mnd_ew<<<4096, 256, 0, stream>>>(rows, cols, ods + (size_t)r * N_, ids + (size_t)r * N_, ewr + (size_t)r * E_);
  }

  // ---- one conv layer invocation ----
  auto conv = [&](int l, const float* hin, bool pos, float* ph_slot, float* g_slot) {
    mnd_zero<<<4096, 256, 0, stream>>>(outacc, (long)NPAD_ * D_);
    mnd_zero<<<1, 256, 0, stream>>>(bnsum, (long)D_);
    mnd_zero<<<1, 256, 0, stream>>>(bnsq, (long)D_);
    mnd_l1<<<NW, 256, 0, stream>>>(hin, invl1);
    mnd_cvt_bf<<<4096, 256, 0, stream>>>(hin, hbf, (long)N_ * D_);
    for (int r = 0; r < R_; ++r) {
      const int* rows = ei + (size_t)r * 2 * E_;
      const int* cols = rows + E_;
      mnd_zero<<<4096, 256, 0, stream>>>(Sall, 6L * N_ * D_);
      mnd_spmm6<<<20000, 256, 0, stream>>>(rows, cols, ewr + (size_t)r * E_, hin, invl1,
                                           ShO, SxO, SxxO, ShI, SxI, SxxI);
      mnd_cvt_bf<<<4096, 256, 0, stream>>>(ShO, ShObf, (long)N_ * D_);
      mnd_cvt_bf<<<4096, 256, 0, stream>>>(ShI, ShIbf, (long)N_ * D_);
      mnd_cfeat<<<NW, 256, 0, stream>>>(hin, invl1, SxO, SxxO, SxI, SxxI,
          out_emb + (size_t)l * MAXDEG_ * D_, in_emb + (size_t)l * MAXDEG_ * D_,
          idxO + (size_t)r * N_, idxI + (size_t)r * N_,
          W_outf + (size_t)l * D_, W_inf + (size_t)l * D_,
          b_outf + l, b_inf + l, tau_p,
          out_mask + (size_t)r * N_, out_mb + (size_t)r * N_,
          in_mask + (size_t)r * N_, in_mb + (size_t)r * N_,
          co_arr, ci_arr,
          pos ? o_cout : nullptr, pos ? o_cin : nullptr, 1.0f / (L_ * R_));
      mnd_gemm_scaled<<<GB_, 256, 0, stream>>>(ShObf, WsrcT + (size_t)l * D_ * D_,
                                               b_src + (size_t)l * D_, co_arr, outacc);
      mnd_gemm_scaled<<<GB_, 256, 0, stream>>>(ShIbf, WdstT + (size_t)l * D_ * D_,
                                               b_dst + (size_t)l * D_, ci_arr, outacc);
    }
    mnd_gemm_fc<<<GB_, 256, 0, stream>>>(hbf, WfcT + (size_t)l * D_ * D_,
                                         b_fc + (size_t)l * D_, alpha_p, outacc);
    mnd_bnstats<<<256, 256, 0, stream>>>(outacc, bnsum, bnsq);
    mnd_bn<<<4096, 256, 0, stream>>>(outacc, bnsum, bnsq,
                                     bn_gamma + (size_t)l * D_, bn_beta + (size_t)l * D_,
                                     ph_slot, g_slot);
  };

  // ---- layers ----
  conv(0, x, true, o_ph0, o_g0);
  mnd_perm<<<4096, 256, 0, stream>>>(x, hnegbuf, 100u);
  conv(0, hnegbuf, false, o_nh0, nullptr);
  conv(1, o_ph0, true, o_ph1, o_g1);
  mnd_perm<<<4096, 256, 0, stream>>>(o_nh0, hnegbuf, 101u);
  conv(1, hnegbuf, false, o_nh1, nullptr);

  // ---- JK-max + classifier + log-softmax ----
  mnd_jkmax<<<4096, 256, 0, stream>>>(o_ph0, o_ph1, jkbf);
  mnd_gemm_lin<<<GB_, 256, 0, stream>>>(jkbf, WlinT, b_lin, logits);
  mnd_logsm<<<NW, 256, 0, stream>>>(logits, o_logp);
}
